// FSUMatMul_1022202217407
// MI455X (gfx1250) — 8-bit matrix operand order repaired and checked by reading, compiled, not run on hardware
//
#include <hip/hip_runtime.h>
#include <hip/hip_bf16.h>
#include <stdint.h>

typedef __attribute__((ext_vector_type(16))) int   v16i;
typedef __attribute__((ext_vector_type(8)))  float v8f;

#define IN_F   2048
#define BATCH  4096
#define OUT_F  4096
#define WIDTH_SCALE 256.0f
#define DEPTH_MAX   4095.0f
#define FP8_ONE 0x38u

#define KCHUNK     128
#define NCHUNKS    (IN_F / KCHUNK)
#define LDS_STRIDE 144

__global__ __launch_bounds__(256) void pack_x_kernel(const float* __restrict__ x,
                                                     unsigned char* __restrict__ Apk,
                                                     float* __restrict__ sx) {
    const int row = blockIdx.x;
    const float* src = x + (size_t)row * IN_F;
    unsigned char* dst = Apk + (size_t)row * IN_F;
    __shared__ float red[256];

    const int base = threadIdx.x * 8;
    float vals[8];
    *(float4*)(vals)     = *(const float4*)(src + base);
    *(float4*)(vals + 4) = *(const float4*)(src + base + 4);

    float s = 0.0f;
    unsigned int lo = 0, hi = 0;
#pragma unroll
    for (int i = 0; i < 8; ++i) {
        s += vals[i];
        unsigned int byte = (vals[i] != 0.0f) ? FP8_ONE : 0u;
        if (i < 4) lo |= byte << (8 * i);
        else       hi |= byte << (8 * (i - 4));
    }
    uint2 packed; packed.x = lo; packed.y = hi;
    *(uint2*)(dst + base) = packed;

    red[threadIdx.x] = s;
    __syncthreads();
    for (int off = 128; off > 0; off >>= 1) {
        if (threadIdx.x < off) red[threadIdx.x] += red[threadIdx.x + off];
        __syncthreads();
    }
    if (threadIdx.x == 0) sx[row] = red[0];
}

__global__ __launch_bounds__(256) void pack_w_kernel(const float* __restrict__ w,
                                                     const float* __restrict__ rng_val,
                                                     unsigned char* __restrict__ Bpk,
                                                     float* __restrict__ sp) {
    const int row = blockIdx.x;
    const float rng = rng_val[0];
    const float* src = w + (size_t)row * IN_F;
    unsigned char* dst = Bpk + (size_t)row * IN_F;
    __shared__ float red[256];

    const int base = threadIdx.x * 8;
    float vals[8];
    *(float4*)(vals)     = *(const float4*)(src + base);
    *(float4*)(vals + 4) = *(const float4*)(src + base + 4);

    float s = 0.0f;
    unsigned int lo = 0, hi = 0;
#pragma unroll
    for (int i = 0; i < 8; ++i) {
        float source = rintf((vals[i] + 1.0f) * 0.5f * WIDTH_SCALE);
        unsigned int bit = (source > rng) ? 1u : 0u;
        s += (float)bit;
        unsigned int byte = bit * FP8_ONE;
        if (i < 4) lo |= byte << (8 * i);
        else       hi |= byte << (8 * (i - 4));
    }
    uint2 packed; packed.x = lo; packed.y = hi;
    *(uint2*)(dst + base) = packed;

    red[threadIdx.x] = s;
    __syncthreads();
    for (int off = 128; off > 0; off >>= 1) {
        if (threadIdx.x < off) red[threadIdx.x] += red[threadIdx.x + off];
        __syncthreads();
    }
    if (threadIdx.x == 0) sp[row] = red[0];
}

__global__ __launch_bounds__(256) void fsu_wmma_gemm(const unsigned char* __restrict__ Apk,
                                                     const unsigned char* __restrict__ Bpk,
                                                     const float* __restrict__ sx,
                                                     const float* __restrict__ sp,
                                                     float* __restrict__ out) {
    __shared__ __align__(16) unsigned char bbuf[2][64 * LDS_STRIDE];

    const int tid  = threadIdx.x;
    const int wave = tid >> 5;
    const int lane = tid & 31;

    const int nBlock = blockIdx.x & 63;
    const int mBlock = blockIdx.x >> 6;
    const int tNbase = nBlock << 6;
    const int mBase  = mBlock * 256 + wave * 32;

    const int nl  = lane & 15;
    const int h   = lane >> 4;
    const int h8  = h * 8;
    const int h16 = h * 16;

    const unsigned char* arow0 = Apk + (size_t)(mBase + nl) * IN_F;
    const unsigned char* arow1 = arow0 + (size_t)16 * IN_F;

    const int srow = tid >> 2;
    const int sq   = (tid & 3) * 32;
    const unsigned char* sgl = Bpk + (size_t)(tNbase + srow) * IN_F + sq;

    v8f acc[2][4] = {};

    {
        uint4 d0 = *(const uint4*)(sgl);
        uint4 d1 = *(const uint4*)(sgl + 16);
        unsigned char* dst = &bbuf[0][srow * LDS_STRIDE + sq];
        *(uint4*)(dst)      = d0;
        *(uint4*)(dst + 16) = d1;
    }
    __syncthreads();

    for (int chunk = 0; chunk < NCHUNKS; ++chunk) {
        const int cur = chunk & 1;
        const int k0  = chunk * KCHUNK;

        if (chunk + 1 < NCHUNKS) {
            uint4 d0 = *(const uint4*)(sgl + k0 + KCHUNK);
            uint4 d1 = *(const uint4*)(sgl + k0 + KCHUNK + 16);
            unsigned char* dst = &bbuf[cur ^ 1][srow * LDS_STRIDE + sq];
            *(uint4*)(dst)      = d0;
            *(uint4*)(dst + 16) = d1;
        }

        uint2 a0[8], a1[8];
#pragma unroll
        for (int j = 0; j < 8; ++j) {
            a0[j] = *(const uint2*)(arow0 + k0 + (j >> 1) * 32 + h16 + (j & 1) * 8);
            a1[j] = *(const uint2*)(arow1 + k0 + (j >> 1) * 32 + h16 + (j & 1) * 8);
        }
        v16i A0 = { (int)a0[0].x, (int)a0[0].y, (int)a0[1].x, (int)a0[1].y,
                    (int)a0[2].x, (int)a0[2].y, (int)a0[3].x, (int)a0[3].y,
                    (int)a0[4].x, (int)a0[4].y, (int)a0[5].x, (int)a0[5].y,
                    (int)a0[6].x, (int)a0[6].y, (int)a0[7].x, (int)a0[7].y };
        v16i A1 = { (int)a1[0].x, (int)a1[0].y, (int)a1[1].x, (int)a1[1].y,
                    (int)a1[2].x, (int)a1[2].y, (int)a1[3].x, (int)a1[3].y,
                    (int)a1[4].x, (int)a1[4].y, (int)a1[5].x, (int)a1[5].y,
                    (int)a1[6].x, (int)a1[6].y, (int)a1[7].x, (int)a1[7].y };

#pragma unroll
        for (int t = 0; t < 4; ++t) {
            const unsigned char* bsrc = &bbuf[cur][(t * 16 + nl) * LDS_STRIDE + h16];
            uint4 b0 = *(const uint4*)(bsrc);
            uint4 b1 = *(const uint4*)(bsrc + 32);
            uint4 b2 = *(const uint4*)(bsrc + 64);
            uint4 b3 = *(const uint4*)(bsrc + 96);
            v16i B = { (int)b0.x, (int)b0.y, (int)b0.z, (int)b0.w,
                       (int)b1.x, (int)b1.y, (int)b1.z, (int)b1.w,
                       (int)b2.x, (int)b2.y, (int)b2.z, (int)b2.w,
                       (int)b3.x, (int)b3.y, (int)b3.z, (int)b3.w };
            acc[0][t] = __builtin_amdgcn_wmma_f32_16x16x128_fp8_fp8(A0, B, (short)0, acc[0][t], false, false);
            acc[1][t] = __builtin_amdgcn_wmma_f32_16x16x128_fp8_fp8(A1, B, (short)0, acc[1][t], false, false);
        }

        __syncthreads();
    }

    float sxr[2][8];
#pragma unroll
    for (int f = 0; f < 2; ++f)
#pragma unroll
        for (int r = 0; r < 8; ++r)
            sxr[f][r] = sx[mBase + f * 16 + r + 8 * h];

    const float inf_f = (float)IN_F;

#pragma unroll
    for (int t = 0; t < 4; ++t) {
        const int o = tNbase + t * 16 + nl;
        const float spv = sp[o];
#pragma unroll
        for (int f = 0; f < 2; ++f) {
#pragma unroll
            for (int r = 0; r < 8; ++r) {
                const int gm = mBase + f * 16 + r + 8 * h;
                float sum_prod = inf_f - sxr[f][r] - spv + 2.0f * acc[f][t][r];
                float clipped  = fminf(fmaxf(sum_prod, 0.0f), DEPTH_MAX);
                out[(size_t)gm * OUT_F + o] = (clipped >= inf_f) ? 1.0f : 0.0f;
            }
        }
    }
}

extern "C" void kernel_launch(void* const* d_in, const int* in_sizes, int n_in,
                              void* d_out, int out_size, void* d_ws, size_t ws_size,
                              hipStream_t stream) {
    const float* x   = (const float*)d_in[0];
    const float* w   = (const float*)d_in[1];
    const float* rng = (const float*)d_in[2];
    float* out = (float*)d_out;

    char* ws = (char*)d_ws;
    unsigned char* Apk = (unsigned char*)ws;
    unsigned char* Bpk = (unsigned char*)(ws + (size_t)BATCH * IN_F);
    float* sx = (float*)(ws + 2 * (size_t)BATCH * IN_F);
    float* sp = (float*)(ws + 2 * (size_t)BATCH * IN_F + BATCH * sizeof(float));

    pack_x_kernel<<<BATCH, 256, 0, stream>>>(x, Apk, sx);
    pack_w_kernel<<<OUT_F, 256, 0, stream>>>(w, rng, Bpk, sp);

    const int nblocks = 16 * 64;
    fsu_wmma_gemm<<<nblocks, 256, 0, stream>>>(Apk, Bpk, sx, sp, out);
}
